// DeepSeekDenseLayer_10445360463986
// MI455X (gfx1250) — matrix operand order repaired and checked by reading, not compiled, not run on hardware
//
#include <hip/hip_runtime.h>
#include <hip/hip_bf16.h>

// ---------------- problem constants ----------------
#define B_      4
#define S_      2048
#define D_      2048
#define H_      16
#define QLORA   1536
#define KVLORA  512
#define DNOPE   128
#define DROPE   64
#define DV      128
#define MLP_    8192
#define NT      (B_*S_)          // 8192 tokens
#define DQK     (DNOPE+DROPE)    // 192
#define KVDIM   (KVLORA+DROPE)   // 576
#define HDV     (H_*DV)          // 2048
#define HDQK    (H_*DQK)         // 3072
#define HKVU    (H_*(DNOPE+DV))  // 4096

typedef __attribute__((ext_vector_type(8)))  __bf16 v8bf;
typedef __attribute__((ext_vector_type(16))) __bf16 v16bf;
typedef __attribute__((ext_vector_type(8)))  float  v8f;
typedef __attribute__((ext_vector_type(4)))  unsigned int u32x4;
typedef __attribute__((ext_vector_type(8)))  int i32x8;
typedef __attribute__((ext_vector_type(4)))  int i32x4;

#if __has_builtin(__builtin_amdgcn_tensor_load_to_lds) && __has_builtin(__builtin_amdgcn_s_wait_tensorcnt)
#define HAVE_TDM 1
#else
#define HAVE_TDM 0
#endif

// fp32 -> bf16 bits, round-to-nearest-even
__device__ __forceinline__ unsigned short f2bf(float f) {
    union { float f; unsigned u; } x; x.f = f;
    unsigned r = x.u + 0x7FFFu + ((x.u >> 16) & 1u);
    return (unsigned short)(r >> 16);
}

__device__ __forceinline__ v16bf frag16(const unsigned short* p0, const unsigned short* p1) {
    v8bf lo = *reinterpret_cast<const v8bf*>(p0);
    v8bf hi = *reinterpret_cast<const v8bf*>(p1);
    return __builtin_shufflevector(lo, hi, 0,1,2,3,4,5,6,7,8,9,10,11,12,13,14,15);
}
// A fragment (16x32 bf16): lane row = lane&15; K chunks [h*8..+7] and [h*8+16..+23]
__device__ __forceinline__ v16bf a_frag(const unsigned short* row, int lane_hi) {
    const unsigned short* p = row + lane_hi * 8;
    return frag16(p, p + 16);
}
// B fragment (32x16 bf16) from Bt[n][k]: lane col = lane&15; K chunks [h*8..+7] and [h*8+16..+23]
__device__ __forceinline__ v16bf b_frag(const unsigned short* row, int lane_hi) {
    const unsigned short* p = row + lane_hi * 8;
    return frag16(p, p + 16);
}
__device__ __forceinline__ v8f wmma_bf16(v16bf a, v16bf b, v8f c) {
    return __builtin_amdgcn_wmma_f32_16x16x32_bf16(false, a, false, b, (short)0, c, false, false);
}

#if HAVE_TDM
// Issue one TDM 2D tile load: 64 rows x 32 bf16 elems (64B), global row stride
// `stride_elems`, into LDS at byte offset lds_off with 80B row pitch
// (pad_enable: +4 DWORDs after every 16 DWORDs stored).
__device__ __forceinline__ void tdm_tile_load(unsigned lds_off, const unsigned short* g,
                                              long stride_elems)
{
    unsigned long long ga = (unsigned long long)(uintptr_t)g;
    u32x4 g0;
    g0[0] = 1u;                                            // count=1, user mode
    g0[1] = lds_off;                                       // LDS byte address
    g0[2] = (unsigned)(ga & 0xFFFFFFFFu);                  // global_addr[31:0]
    g0[3] = (unsigned)((ga >> 32) & 0x1FFFFFFu) | (2u << 30); // addr[56:32] | type=2
    i32x8 g1;
    g1[0] = (int)((1u << 16) |            // data_size = 2B
                  (1u << 20) |            // pad_enable
                  (3u << 22) |            // pad_interval: 16 DWORDs
                  (3u << 25));            // pad_amount:   4 DWORDs
    g1[1] = (int)(32u << 16);                              // tensor_dim0 = 32 (lo16)
    g1[2] = (int)(64u << 16);                              // dim0 hi=0 | tensor_dim1=64
    g1[3] = (int)(32u << 16);                              // dim1 hi=0 | tile_dim0=32
    g1[4] = (int)(64u);                                    // tile_dim1=64, tile_dim2=0
    g1[5] = (int)(stride_elems & 0xFFFFFFFFL);             // tensor_dim0_stride lo
    g1[6] = (int)((stride_elems >> 32) & 0xFFFFL);         // stride hi | dim1_stride=0
    g1[7] = 0;
    i32x4 z = {0, 0, 0, 0};
#if __has_include(<hip/amd_detail/amd_gfx1250_TDM.h>)
    i32x8 z8 = {0,0,0,0,0,0,0,0};
    __builtin_amdgcn_tensor_load_to_lds(g0, g1, z, z, z8, 0);
#else
    __builtin_amdgcn_tensor_load_to_lds(g0, g1, z, z, 0);
#endif
}
#endif

// ---------------- bf16 WMMA GEMM: C = A @ Bt^T (+R) ----------------
// A: bf16 [M,K] row-major; Bt: bf16 [N,K] (pre-transposed weights); C fp32.
// grid = (N/64, M/64), block = 128 (4 waves). TDM double-buffered staging.
template<int ADD_RES>
__global__ __launch_bounds__(128)
void gemm64(const unsigned short* __restrict__ A, const unsigned short* __restrict__ Bt,
            const float* __restrict__ R, float* __restrict__ C,
            int K, int lda, int ldbt, int ldc)
{
    __shared__ unsigned short As[2][64][40];   // 80B pitch, 16B aligned rows
    __shared__ unsigned short Bs[2][64][40];
    const int tid  = threadIdx.x;
    const int lane = tid & 31, wv = tid >> 5;
    const int lo16 = lane & 15, hi = lane >> 4;
    const long bm = (long)blockIdx.y * 64, bn = (long)blockIdx.x * 64;

    v8f acc[4] = {};

#if HAVE_TDM
    if (wv == 0) {
        tdm_tile_load((unsigned)(uintptr_t)&As[0][0][0], A + bm * (long)lda, lda);
        tdm_tile_load((unsigned)(uintptr_t)&Bs[0][0][0], Bt + bn * (long)ldbt, ldbt);
    }
#else
    for (int id = tid; id < 256; id += 128) {          // 256 x 16B chunks
        int r = id >> 2, cc = (id & 3) * 8;
        *(uint4*)&As[0][r][cc] = *(const uint4*)(A + (bm + r) * (long)lda + cc);
        *(uint4*)&Bs[0][r][cc] = *(const uint4*)(Bt + (bn + r) * (long)ldbt + cc);
    }
#endif

    int buf = 0;
    for (int k0 = 0; k0 < K; k0 += 32) {
        const bool have_next = (k0 + 32 < K);
        const int nbuf = buf ^ 1;
#if HAVE_TDM
        if (wv == 0) {
            if (have_next) {
                tdm_tile_load((unsigned)(uintptr_t)&As[nbuf][0][0],
                              A + bm * (long)lda + (k0 + 32), lda);
                tdm_tile_load((unsigned)(uintptr_t)&Bs[nbuf][0][0],
                              Bt + bn * (long)ldbt + (k0 + 32), ldbt);
                __builtin_amdgcn_s_wait_tensorcnt(2);  // current tile's 2 ops done
            } else {
                __builtin_amdgcn_s_wait_tensorcnt(0);
            }
        }
#else
        if (have_next) {
            for (int id = tid; id < 256; id += 128) {
                int r = id >> 2, cc = (id & 3) * 8;
                *(uint4*)&As[nbuf][r][cc] =
                    *(const uint4*)(A + (bm + r) * (long)lda + (k0 + 32) + cc);
                *(uint4*)&Bs[nbuf][r][cc] =
                    *(const uint4*)(Bt + (bn + r) * (long)ldbt + (k0 + 32) + cc);
            }
        }
#endif
        __syncthreads();

        v16bf af = a_frag(&As[buf][wv * 16 + lo16][0], hi);
        #pragma unroll
        for (int t = 0; t < 4; ++t) {
            v16bf bf = b_frag(&Bs[buf][t * 16 + lo16][0], hi);
            acc[t] = wmma_bf16(af, bf, acc[t]);
        }
        __syncthreads();
        buf = nbuf;
    }

    const long rbase = bm + wv * 16 + hi * 8;
    #pragma unroll
    for (int t = 0; t < 4; ++t) {
        const long col = bn + t * 16 + lo16;
        #pragma unroll
        for (int i = 0; i < 8; ++i) {
            long row = rbase + i;
            float v = acc[t][i];
            if (ADD_RES) v += R[row * (long)ldc + col];
            C[row * (long)ldc + col] = v;
        }
    }
}

// ---------------- weight convert + transpose: fp32 [K][N] -> bf16 [N][K] --------
__global__ __launch_bounds__(256)
void convt_k(const float* __restrict__ W, unsigned short* __restrict__ Wt, int K, int N)
{
    __shared__ float tile[32][33];
    const int bx = blockIdx.x * 32;            // N
    const int by = blockIdx.y * 32;            // K
    const int tx = threadIdx.x & 31, ty = threadIdx.x >> 5;
    #pragma unroll
    for (int r = ty; r < 32; r += 8)
        tile[r][tx] = W[(long)(by + r) * N + bx + tx];
    __syncthreads();
    #pragma unroll
    for (int r = ty; r < 32; r += 8)
        Wt[(long)(bx + r) * K + by + tx] = f2bf(tile[tx][r]);
}

// ---------------- RMSNorm: fp32 in, bf16 out ----------------
__global__ __launch_bounds__(256)
void rmsnorm_k(const float* __restrict__ x, long ldx, const float* __restrict__ sc,
               unsigned short* __restrict__ y, long ldy, int dim)
{
    const long row = blockIdx.x;
    const float* xr = x + row * ldx;
    float ss = 0.f;
    for (int i = threadIdx.x; i < dim; i += 256) { float v = xr[i]; ss += v * v; }
    #pragma unroll
    for (int m = 16; m >= 1; m >>= 1) ss += __shfl_xor(ss, m, 32);
    __shared__ float red[8];
    if ((threadIdx.x & 31) == 0) red[threadIdx.x >> 5] = ss;
    __syncthreads();
    float tot = 0.f;
    #pragma unroll
    for (int w = 0; w < 8; ++w) tot += red[w];
    float inv = rsqrtf(tot / (float)dim + 1e-6f);
    unsigned short* yr = y + row * ldy;
    for (int i = threadIdx.x; i < dim; i += 256) yr[i] = f2bf(xr[i] * inv * sc[i]);
}

// ---------------- RoPE ----------------
__global__ __launch_bounds__(256)
void rope_q_k(float* __restrict__ q, const int* __restrict__ pos)
{
    const long t = blockIdx.x;
    const float p = (float)pos[t];
    for (int idx = threadIdx.x; idx < H_ * 32; idx += 256) {
        int h = idx >> 5, j = idx & 31;
        float invf = __powf(10000.0f, -(float)(2 * j) / 64.0f);
        float sn, cs; __sincosf(p * invf, &sn, &cs);
        float* base = q + t * (long)HDQK + h * DQK + DNOPE;
        float x1 = base[j], x2 = base[j + 32];
        base[j]      = x1 * cs - x2 * sn;
        base[j + 32] = x2 * cs + x1 * sn;
    }
}

__global__ __launch_bounds__(64)
void rope_k_k(float* __restrict__ kv, const int* __restrict__ pos)
{
    const long t = blockIdx.x;
    const int j = threadIdx.x;
    if (j >= 32) return;
    const float p = (float)pos[t];
    float invf = __powf(10000.0f, -(float)(2 * j) / 64.0f);
    float sn, cs; __sincosf(p * invf, &sn, &cs);
    float* base = kv + t * (long)KVDIM + KVLORA;
    float x1 = base[j], x2 = base[j + 32];
    base[j]      = x1 * cs - x2 * sn;
    base[j + 32] = x2 * cs + x1 * sn;
}

// kf[t][h][0:128] = k_nope (kvu), kf[t][h][128:192] = roped shared k_rope
__global__ __launch_bounds__(256)
void build_kf_k(const float* __restrict__ kvu, const float* __restrict__ kvb,
                float* __restrict__ kf)
{
    const long t = blockIdx.x;
    for (int idx = threadIdx.x; idx < HDQK; idx += 256) {
        int h = idx / DQK, d = idx % DQK;
        float v = (d < DNOPE) ? kvu[t * (long)HKVU + h * (DNOPE + DV) + d]
                              : kvb[t * (long)KVDIM + KVLORA + (d - DNOPE)];
        kf[t * (long)HDQK + idx] = v;
    }
}

// ---------------- flash attention (causal; positions arange, segs uniform) -------
// grid = (S/64, H, B), block = 128 (4 waves); wave owns 16 query rows; bf16 out.
__global__ __launch_bounds__(128)
void flash_k(const float* __restrict__ q, const float* __restrict__ kf,
             const float* __restrict__ kvu, unsigned short* __restrict__ attn)
{
    __shared__ unsigned short Qs[64][200];     // [q][d] bf16, prescaled
    __shared__ unsigned short Ks[64][200];     // [k][d] bf16
    __shared__ unsigned short Vt[128][72];     // [d][k] bf16
    __shared__ unsigned short Ps[4][16][72];   // per-wave probs [row][key]

    const int qb = blockIdx.x, h = blockIdx.y, b = blockIdx.z;
    const int tid = threadIdx.x, lane = tid & 31, wv = tid >> 5;
    const int lo16 = lane & 15, hi = lane >> 4;
    const long tok0 = (long)b * S_;
    const float qscale = 0.0721687836f;   // 1/sqrt(192)

    for (int i = tid * 4; i < 64 * DQK; i += 512) {
        int r = i / DQK, c = i % DQK;
        const float4 f = *reinterpret_cast<const float4*>(
            q + (tok0 + qb * 64 + r) * (long)HDQK + h * DQK + c);
        Qs[r][c+0] = f2bf(f.x * qscale); Qs[r][c+1] = f2bf(f.y * qscale);
        Qs[r][c+2] = f2bf(f.z * qscale); Qs[r][c+3] = f2bf(f.w * qscale);
    }

    v8f acc[8] = {};
    float mrow[8], lrow[8];
    #pragma unroll
    for (int i = 0; i < 8; ++i) { mrow[i] = -1e30f; lrow[i] = 0.f; }

    for (int kb = 0; kb <= qb; ++kb) {
        for (int i = tid * 4; i < 64 * DQK; i += 512) {
            int r = i / DQK, c = i % DQK;
            const float4 f = *reinterpret_cast<const float4*>(
                kf + (tok0 + kb * 64 + r) * (long)HDQK + h * DQK + c);
            Ks[r][c+0] = f2bf(f.x); Ks[r][c+1] = f2bf(f.y);
            Ks[r][c+2] = f2bf(f.z); Ks[r][c+3] = f2bf(f.w);
        }
        for (int i = tid * 4; i < 64 * DV; i += 512) {
            int r = i / DV, c = i % DV;
            const float4 f = *reinterpret_cast<const float4*>(
                kvu + (tok0 + kb * 64 + r) * (long)HKVU + h * (DNOPE + DV) + DNOPE + c);
            Vt[c+0][r] = f2bf(f.x); Vt[c+1][r] = f2bf(f.y);
            Vt[c+2][r] = f2bf(f.z); Vt[c+3][r] = f2bf(f.w);
        }
        __syncthreads();

        v8f sc[4];
        #pragma unroll
        for (int t = 0; t < 4; ++t) {
            v8f s = {};
            #pragma unroll
            for (int ks = 0; ks < 6; ++ks) {
                v16bf af = a_frag(&Qs[wv * 16 + lo16][ks * 32], hi);
                v16bf bf = b_frag(&Ks[t * 16 + lo16][ks * 32], hi);
                s = wmma_bf16(af, bf, s);
            }
            sc[t] = s;
        }

        if (kb == qb) {
            #pragma unroll
            for (int t = 0; t < 4; ++t)
                #pragma unroll
                for (int i = 0; i < 8; ++i) {
                    int qi = wv * 16 + hi * 8 + i;
                    int ki = t * 16 + lo16;
                    if (ki > qi) sc[t][i] = -1e9f;
                }
        }

        #pragma unroll
        for (int i = 0; i < 8; ++i) {
            float rmax = fmaxf(fmaxf(sc[0][i], sc[1][i]), fmaxf(sc[2][i], sc[3][i]));
            #pragma unroll
            for (int m = 8; m >= 1; m >>= 1) rmax = fmaxf(rmax, __shfl_xor(rmax, m, 32));
            float mnew  = fmaxf(mrow[i], rmax);
            float alpha = __expf(mrow[i] - mnew);
            mrow[i] = mnew;
            float rsum = 0.f;
            #pragma unroll
            for (int t = 0; t < 4; ++t) {
                float p = __expf(sc[t][i] - mnew);
                sc[t][i] = p;
                rsum += p;
            }
            #pragma unroll
            for (int m = 8; m >= 1; m >>= 1) rsum += __shfl_xor(rsum, m, 32);
            lrow[i] = lrow[i] * alpha + rsum;
            #pragma unroll
            for (int t = 0; t < 8; ++t) acc[t][i] *= alpha;
        }

        #pragma unroll
        for (int t = 0; t < 4; ++t)
            #pragma unroll
            for (int i = 0; i < 8; ++i)
                Ps[wv][hi * 8 + i][t * 16 + lo16] = f2bf(sc[t][i]);

        #pragma unroll
        for (int ks = 0; ks < 2; ++ks) {
            v16bf af = a_frag(&Ps[wv][lo16][ks * 32], hi);
            #pragma unroll
            for (int t = 0; t < 8; ++t) {
                v16bf bf = b_frag(&Vt[t * 16 + lo16][ks * 32], hi);
                acc[t] = wmma_bf16(af, bf, acc[t]);
            }
        }
        __syncthreads();
    }

    #pragma unroll
    for (int t = 0; t < 8; ++t)
        #pragma unroll
        for (int i = 0; i < 8; ++i) {
            long row = tok0 + qb * 64 + wv * 16 + hi * 8 + i;
            long col = (long)h * DV + t * 16 + lo16;
            attn[row * (long)HDV + col] = f2bf(acc[t][i] / lrow[i]);
        }
}

// ---------------- SiLU gate: fp32 g,u -> bf16 act ----------------
__global__ __launch_bounds__(256)
void silu_mul_k(const float* __restrict__ g, const float* __restrict__ u,
                unsigned short* __restrict__ act, long n)
{
    long i = (long)blockIdx.x * 256 + threadIdx.x;
    if (i < n) { float x = g[i]; act[i] = f2bf((x / (1.f + __expf(-x))) * u[i]); }
}

// ---------------- host launcher ----------------
extern "C" void kernel_launch(void* const* d_in, const int* in_sizes, int n_in,
                              void* d_out, int out_size, void* d_ws, size_t ws_size,
                              hipStream_t stream)
{
    const float* inputs  = (const float*)d_in[0];
    const int*   pos     = (const int*)  d_in[2];
    const float* pre_ln  = (const float*)d_in[3];
    const float* post_ln = (const float*)d_in[4];
    const float* q_ln    = (const float*)d_in[5];
    const float* kv_ln   = (const float*)d_in[6];
    const float* wq_a    = (const float*)d_in[7];
    const float* wq_b    = (const float*)d_in[8];
    const float* wkv_a   = (const float*)d_in[9];
    const float* wkv_b   = (const float*)d_in[10];
    const float* wo_at   = (const float*)d_in[11];
    const float* wi0     = (const float*)d_in[12];
    const float* wi1     = (const float*)d_in[13];
    const float* wo_mlp  = (const float*)d_in[14];
    float* out = (float*)d_out;
    (void)in_sizes; (void)n_in; (void)out_size; (void)ws_size;

    // -------- fp32 scratch --------
    float* wsf = (float*)d_ws;
    size_t off = 0;
    auto allocf = [&](size_t n) { float* p = wsf + off; off += n; return p; };
    float* qc    = allocf((size_t)NT * QLORA);   // region reused as g after flash
    float* qbuf  = allocf((size_t)NT * HDQK);
    float* kvb   = allocf((size_t)NT * KVDIM);
    float* kvu   = allocf((size_t)NT * HKVU);
    float* kfb   = allocf((size_t)NT * HDQK);
    float* inter = allocf((size_t)NT * D_);
    float* ubuf  = allocf((size_t)NT * MLP_);
    float* gbuf  = qc;   // qc/qbuf/kvb/kvu/kfb (101M floats) all dead after flash

    // -------- bf16 scratch --------
    unsigned short* wsh = (unsigned short*)(wsf + off);
    size_t hoff = 0;
    auto alloch = [&](size_t n) { unsigned short* p = wsh + hoff; hoff += n; return p; };
    unsigned short* w_qa  = alloch((size_t)QLORA * D_);
    unsigned short* w_qb  = alloch((size_t)HDQK * QLORA);
    unsigned short* w_kva = alloch((size_t)KVDIM * D_);
    unsigned short* w_kvb = alloch((size_t)HKVU * KVLORA);
    unsigned short* w_o   = alloch((size_t)D_ * HDV);
    unsigned short* w_i0  = alloch((size_t)MLP_ * D_);
    unsigned short* w_i1  = alloch((size_t)MLP_ * D_);
    unsigned short* w_om  = alloch((size_t)D_ * MLP_);
    unsigned short* lnx   = alloch((size_t)NT * D_);
    unsigned short* qcn   = alloch((size_t)NT * QLORA);
    unsigned short* ckv   = alloch((size_t)NT * KVLORA);
    unsigned short* attn  = alloch((size_t)NT * HDV);
    unsigned short* hid   = alloch((size_t)NT * D_);
    unsigned short* act   = alloch((size_t)NT * MLP_);

    // 0) convert + transpose all weights to bf16 [N][K]
    convt_k<<<dim3(QLORA/32, D_/32),   256, 0, stream>>>(wq_a,   w_qa,  D_,    QLORA);
    convt_k<<<dim3(HDQK/32,  QLORA/32),256, 0, stream>>>(wq_b,   w_qb,  QLORA, HDQK);
    convt_k<<<dim3(KVDIM/32, D_/32),   256, 0, stream>>>(wkv_a,  w_kva, D_,    KVDIM);
    convt_k<<<dim3(HKVU/32,  KVLORA/32),256,0, stream>>>(wkv_b,  w_kvb, KVLORA,HKVU);
    convt_k<<<dim3(D_/32,    HDV/32),  256, 0, stream>>>(wo_at,  w_o,   HDV,   D_);
    convt_k<<<dim3(MLP_/32,  D_/32),   256, 0, stream>>>(wi0,    w_i0,  D_,    MLP_);
    convt_k<<<dim3(MLP_/32,  D_/32),   256, 0, stream>>>(wi1,    w_i1,  D_,    MLP_);
    convt_k<<<dim3(D_/32,    MLP_/32), 256, 0, stream>>>(wo_mlp, w_om,  MLP_,  D_);

    // 1) lnx = rmsnorm(inputs) * pre_ln  (bf16)
    rmsnorm_k<<<NT, 256, 0, stream>>>(inputs, D_, pre_ln, lnx, D_, D_);
    // 2) qc = lnx @ wq_a
    gemm64<0><<<dim3(QLORA/64, NT/64), 128, 0, stream>>>(lnx, w_qa, nullptr, qc, D_, D_, D_, QLORA);
    // 3) qcn = rmsnorm(qc) * q_ln  (bf16)
    rmsnorm_k<<<NT, 256, 0, stream>>>(qc, QLORA, q_ln, qcn, QLORA, QLORA);
    // 4) q = qcn @ wq_b
    gemm64<0><<<dim3(HDQK/64, NT/64), 128, 0, stream>>>(qcn, w_qb, nullptr, qbuf, QLORA, QLORA, QLORA, HDQK);
    // 5) kv = lnx @ wkv_a
    gemm64<0><<<dim3(KVDIM/64, NT/64), 128, 0, stream>>>(lnx, w_kva, nullptr, kvb, D_, D_, D_, KVDIM);
    // 6) ckv = rmsnorm(kv[:, :512]) * kv_ln  (bf16)
    rmsnorm_k<<<NT, 256, 0, stream>>>(kvb, KVDIM, kv_ln, ckv, KVLORA, KVLORA);
    // 7) kvu = ckv @ wkv_b
    gemm64<0><<<dim3(HKVU/64, NT/64), 128, 0, stream>>>(ckv, w_kvb, nullptr, kvu, KVLORA, KVLORA, KVLORA, HKVU);
    // 8,9) RoPE (fp32, in place)
    rope_q_k<<<NT, 256, 0, stream>>>(qbuf, pos);
    rope_k_k<<<NT, 64, 0, stream>>>(kvb, pos);
    // 10) kf = [k_nope | k_rope]
    build_kf_k<<<NT, 256, 0, stream>>>(kvu, kvb, kfb);
    // 11) flash attention -> attn (bf16)
    flash_k<<<dim3(S_/64, H_, B_), 128, 0, stream>>>(qbuf, kfb, kvu, attn);
    // 12) inter = attn @ wo_attn + inputs
    gemm64<1><<<dim3(D_/64, NT/64), 128, 0, stream>>>(attn, w_o, inputs, inter, HDV, HDV, HDV, D_);
    // 13) hid = rmsnorm(inter) * post_ln  (bf16)
    rmsnorm_k<<<NT, 256, 0, stream>>>(inter, D_, post_ln, hid, D_, D_);
    // 14,15) g = hid @ wi0 ; u = hid @ wi1
    gemm64<0><<<dim3(MLP_/64, NT/64), 128, 0, stream>>>(hid, w_i0, nullptr, gbuf, D_, D_, D_, MLP_);
    gemm64<0><<<dim3(MLP_/64, NT/64), 128, 0, stream>>>(hid, w_i1, nullptr, ubuf, D_, D_, D_, MLP_);
    // 16) act = silu(g) * u  (bf16)
    long n = (long)NT * MLP_;
    silu_mul_k<<<(unsigned)((n + 255) / 256), 256, 0, stream>>>(gbuf, ubuf, act, n);
    // 17) out = act @ wo_mlp + inter
    gemm64<1><<<dim3(D_/64, NT/64), 128, 0, stream>>>(act, w_om, inter, out, MLP_, MLP_, MLP_, D_);
}
